// GeoRipNet_64355789964004
// MI455X (gfx1250) — matrix operand order repaired and checked by reading, not compiled, not run on hardware
//
#include <hip/hip_runtime.h>
#include <hip/hip_bf16.h>
#include <stdint.h>

using bf16 = __hip_bfloat16;

typedef __attribute__((ext_vector_type(16))) __bf16    v16bf;
typedef __attribute__((ext_vector_type(8)))  float     v8f;
typedef __attribute__((ext_vector_type(4)))  unsigned  v4u;

// GCC-style vector + address-space-qualified typedefs matching the async
// builtin prototype: (v4i __device__ *, v4i __shared__ *, Imm, Imm)
typedef int v4i_gcc __attribute__((vector_size(16)));
typedef v4i_gcc __attribute__((address_space(1))) as1_v4i;
typedef v4i_gcc __attribute__((address_space(3))) as3_v4i;

union Frag { v4u u4[2]; v16bf bf; };

// ---------------------------------------------------------------------------
// CDNA5 async global->LDS path (ASYNCcnt-tracked DMA), guarded so compile
// cannot break if this toolchain lacks the builtins.
// ---------------------------------------------------------------------------
#if defined(__has_builtin)
#  if __has_builtin(__builtin_amdgcn_global_load_async_to_lds_b128) && \
      __has_builtin(__builtin_amdgcn_s_wait_asynccnt)
#    define ASYNC_OK 1
#  endif
#endif
#ifndef ASYNC_OK
#  define ASYNC_OK 0
#endif

#if defined(__gfx1250__)
#  if ASYNC_OK
#    warning CDNA5_ASYNC_LDS_PATH_ENABLED
#  else
#    warning CDNA5_ASYNC_LDS_PATH_DISABLED_FALLBACK
#  endif
#endif

__device__ __forceinline__ void async_b128(const bf16* g, bf16* l) {
#if ASYNC_OK
    __builtin_amdgcn_global_load_async_to_lds_b128(
        (as1_v4i*)(unsigned long long)(uintptr_t)g,
        (as3_v4i*)(unsigned)(uintptr_t)l,
        0, 0);
#else
    *(v4u*)l = *(const v4u*)g;
#endif
}
__device__ __forceinline__ void async_wait_le2() {
#if ASYNC_OK
    __builtin_amdgcn_s_wait_asynccnt(2);
#endif
}
__device__ __forceinline__ void async_wait_le0() {
#if ASYNC_OK
    __builtin_amdgcn_s_wait_asynccnt(0);
#endif
}

__device__ __forceinline__ v8f zero8() {
    v8f z;
#pragma unroll
    for (int i = 0; i < 8; ++i) z[i] = 0.f;
    return z;
}

__device__ __forceinline__ v8f wmma_bf16(const Frag& a, const Frag& b, v8f c) {
    return __builtin_amdgcn_wmma_f32_16x16x32_bf16(false, a.bf, false, b.bf,
                                                   (short)0, c, false, false);
}

// ---------------------------------------------------------------------------
// constants
// ---------------------------------------------------------------------------
constexpr int Bn = 2048, Kt = 128, Nn = 5, Dd = 128;
constexpr int Mrows = Bn * Kt;          // 262144
constexpr int NDflat = Nn * Dd;         // 640
constexpr int DFF = 4 * Dd;             // 512

// ---------------------------------------------------------------------------
// prep: fold rank-1 embedding through GAT weights.
// wsPrep: [0..15]={alpha,beta,alpha2,beta2}[4], [16..527]=U[4][128],
//         [528..1039]=V[4][128]
// ---------------------------------------------------------------------------
__global__ __launch_bounds__(128) void prep_kernel(
    const float* __restrict__ we, const float* __restrict__ be,
    const float* __restrict__ gat_W, const float* __restrict__ gat_a,
    const float* __restrict__ gat_out_W, float* __restrict__ wsPrep) {
    __shared__ float u[128], v[128];
    int t = threadIdx.x;
    float su = 0.f, sv = 0.f;
    for (int d = 0; d < 128; ++d) {
        float w = gat_W[d * 128 + t];
        su += we[d] * w;
        sv += be[d] * w;
    }
    u[t] = su; v[t] = sv;
    __syncthreads();
    if (t < 16) {
        int h = t & 3, which = t >> 2;  // 0:alpha 1:beta 2:alpha2 3:beta2
        const float* av  = gat_a + h * 64 + ((which >= 2) ? 32 : 0);
        const float* src = ((which & 1) == 0) ? u : v;
        float s = 0.f;
        for (int d = 0; d < 32; ++d) s += src[h * 32 + d] * av[d];
        wsPrep[which * 4 + h] = s;
    }
    for (int h = 0; h < 4; ++h) {
        float sU = 0.f, sV = 0.f;
        for (int d = 0; d < 32; ++d) {
            float w = gat_out_W[(h * 32 + d) * 128 + t];
            sU += u[h * 32 + d] * w;
            sV += v[h * 32 + d] * w;
        }
        wsPrep[16 + h * 128 + t]        = sU;
        wsPrep[16 + 512 + h * 128 + t]  = sV;
    }
}

// ---------------------------------------------------------------------------
// dynamic adjacency: a_dyn = a_static * sigmoid(gdelt . wg + bg)
// ---------------------------------------------------------------------------
__global__ void adyn_kernel(const float* __restrict__ gdelt,
                            const float* __restrict__ a_static,
                            const float* __restrict__ wg,
                            const float* __restrict__ bg,
                            float* __restrict__ adyn, int total) {
    int i = blockIdx.x * blockDim.x + threadIdx.x;
    if (i >= total) return;
    const float* g = gdelt + (size_t)i * 3;
    float z = g[0] * wg[0] + g[1] * wg[1] + g[2] * wg[2] + bg[0];
    adyn[i] = a_static[i] * (1.f / (1.f + __expf(-z)));
}

// ---------------------------------------------------------------------------
// fused GAT (rank-1 folded): per (b,k) scalar softmax over N=5, then
// Xg[row, i*128+d] = elu( sum_h q_ih*U_h[d] + s_ih*V_h[d] + gat_out_b[d] )
// ---------------------------------------------------------------------------
__global__ __launch_bounds__(128) void gat_kernel(
    const float* __restrict__ prices, const float* __restrict__ adyn,
    const float* __restrict__ wsPrep, const float* __restrict__ gat_out_b,
    bf16* __restrict__ Xg) {
    int bk = blockIdx.x;
    int b  = bk >> 7;
    __shared__ float sp[5], sad[25], sq[20], ssm[20], sab[16];
    int t = threadIdx.x;
    if (t < 5)  sp[t]  = prices[(size_t)bk * 5 + t];
    if (t < 25) sad[t] = adyn[(size_t)b * 25 + t];
    if (t < 16) sab[t] = wsPrep[t];
    __syncthreads();
    if (t < 20) {
        int i = t >> 2, h = t & 3;
        float alpha = sab[h], beta = sab[4 + h], alpha2 = sab[8 + h], beta2 = sab[12 + h];
        float ei = sp[i] * alpha + beta;
        float e[5]; bool msk[5];
        float mx = -1e30f; bool any = false;
        for (int j = 0; j < 5; ++j) {
            bool ok = (sad[i * 5 + j] != 0.f);
            msk[j] = ok;
            float x = ei + sp[j] * alpha2 + beta2;
            x = (x > 0.f) ? x : 0.2f * x;     // leaky_relu(0.2)
            e[j] = x;
            if (ok) { any = true; mx = fmaxf(mx, x); }
        }
        float den = 0.f, num = 0.f;
        for (int j = 0; j < 5; ++j)
            if (msk[j]) { float w = __expf(e[j] - mx); den += w; num += w * sp[j]; }
        sq[t]  = any ? num / den : 0.f;       // nan_to_num: all-masked row -> 0
        ssm[t] = any ? 1.f : 0.f;
    }
    __syncthreads();
    const float* U = wsPrep + 16;
    const float* V = wsPrep + 16 + 512;
    float gb = gat_out_b[t];
    for (int i = 0; i < 5; ++i) {
        float val = gb;
#pragma unroll
        for (int h = 0; h < 4; ++h)
            val += sq[i * 4 + h] * U[h * 128 + t] + ssm[i * 4 + h] * V[h * 128 + t];
        val = (val > 0.f) ? val : (__expf(val) - 1.f);   // elu
        Xg[(size_t)bk * NDflat + i * 128 + t] = __float2bfloat16(val);
    }
}

// ---------------------------------------------------------------------------
// weight convert+transpose: src f32 [R,C] row-major -> dst bf16 [Cpad,R]
// ---------------------------------------------------------------------------
__global__ void wconv_kernel(const float* __restrict__ src, bf16* __restrict__ dst,
                             int R, int C, int Cpad) {
    int idx = blockIdx.x * blockDim.x + threadIdx.x;
    if (idx >= Cpad * R) return;
    int n = idx / R, kk = idx - n * R;
    float v = (n < C) ? src[(size_t)kk * C + n] : 0.f;
    dst[idx] = __float2bfloat16(v);
}

// ---------------------------------------------------------------------------
// WMMA GEMM with async-LDS double-buffered B staging.
// C[M, y*128..] = A[M,Kc](bf16 row-major) x BT[Nc,Kc](bf16)
// 256 threads = 8 waves; wave owns 16 rows x (NT*16) cols; K chunk = 32.
// MODE 0: +bias -> bf16 | 1: +bias relu -> bf16 | 2: +bias+pos_enc -> f32+bf16
// MODE 3: +bias+residual+LayerNorm(128) -> f32+bf16 | 4: +bias f32 cols<outCols
// ---------------------------------------------------------------------------
template <int NT, int MODE>
__global__ __launch_bounds__(256) void gemm_kernel(
    const bf16* __restrict__ A, int ldA, int Kc,
    const bf16* __restrict__ BT,
    const float* __restrict__ bias,
    const float* __restrict__ resid, int ldR,
    const float* __restrict__ lng, const float* __restrict__ lnb,
    const float* __restrict__ posenc,
    float* __restrict__ outF, int ldF,
    bf16* __restrict__ outB, int ldB16, int outCols) {
    __shared__ bf16 btile[2][128 * 32];   // 2 x 8 KB double buffer

    const int tid  = threadIdx.x;
    const int lane = tid & 31;
    const int wave = tid >> 5;
    const int l15  = lane & 15;
    const int half = lane >> 4;
    const int m0   = blockIdx.x * 128 + wave * 16;
    const int nblk = blockIdx.y * 128;

    // cooperative async staging: 2 threads per B^T row, two b128 per half-row
    const int rown = tid >> 1;            // 0..127
    const int seg  = (tid & 1) * 16;      // bf16 offset inside 32-wide chunk
    const bool stager = (rown < NT * 16);
    auto stage = [&](int buf, int k0) {
        if (stager) {
            const bf16* src = BT + (size_t)(nblk + rown) * Kc + k0 + seg;
            bf16* dst = &btile[buf][rown * 32 + seg];
            async_b128(src, dst);
            async_b128(src + 8, dst + 8);
        }
    };

    v8f acc[NT];
#pragma unroll
    for (int t = 0; t < NT; ++t) acc[t] = zero8();

    // A layout (16x32 bf16): lane<16 -> M=lane, contiguous K: k = 16*(lane/16) + s.
    const bf16* Arow = A + (size_t)(m0 + l15) * ldA + 16 * half;

    const int nchunks = Kc / 32;
    stage(0, 0);
    for (int c = 0; c < nchunks; ++c) {
        const int k0 = c * 32;
        if (c + 1 < nchunks) { stage((c + 1) & 1, k0 + 32); async_wait_le2(); }
        else                 { async_wait_le0(); }
        __syncthreads();

        __builtin_prefetch(Arow + k0 + 128, 0, 0);   // global_prefetch_b8
        Frag a;
        a.u4[0] = *(const v4u*)(Arow + k0);
        a.u4[1] = *(const v4u*)(Arow + k0 + 8);
        const bf16* bt = btile[c & 1];
#pragma unroll
        for (int t = 0; t < NT; ++t) {
            const bf16* Brow = bt + (t * 16 + l15) * 32 + 16 * half;
            Frag b;
            b.u4[0] = *(const v4u*)(Brow);
            b.u4[1] = *(const v4u*)(Brow + 8);
            acc[t] = wmma_bf16(a, b, acc[t]);
        }
        __syncthreads();
    }

    // C layout: acc[t][r] = C(m0 + r + 8*half, nblk + t*16 + l15)
    if constexpr (MODE == 3) {
#pragma unroll
        for (int t = 0; t < NT; ++t) {
            int col = nblk + t * 16 + l15;
            float bv = bias[col];
#pragma unroll
            for (int r = 0; r < 8; ++r) {
                int row = m0 + r + 8 * half;
                acc[t][r] += bv + resid[(size_t)row * ldR + col];
            }
        }
        float mean[8], rstd[8];
#pragma unroll
        for (int r = 0; r < 8; ++r) {
            float sA = 0.f, sQ = 0.f;
#pragma unroll
            for (int t = 0; t < NT; ++t) { float v = acc[t][r]; sA += v; sQ += v * v; }
#pragma unroll
            for (int mm = 1; mm < 16; mm <<= 1) {
                sA += __shfl_xor(sA, mm, 32);
                sQ += __shfl_xor(sQ, mm, 32);
            }
            float mu = sA * (1.f / 128.f);
            mean[r] = mu;
            rstd[r] = rsqrtf(sQ * (1.f / 128.f) - mu * mu + 1e-5f);
        }
#pragma unroll
        for (int t = 0; t < NT; ++t) {
            int col = nblk + t * 16 + l15;
            float gg = lng[col], bb = lnb[col];
#pragma unroll
            for (int r = 0; r < 8; ++r) {
                int row = m0 + r + 8 * half;
                float val = (acc[t][r] - mean[r]) * rstd[r] * gg + bb;
                outF[(size_t)row * ldF + col]   = val;
                outB[(size_t)row * ldB16 + col] = __float2bfloat16(val);
            }
        }
    } else {
#pragma unroll
        for (int t = 0; t < NT; ++t) {
            int col = nblk + t * 16 + l15;
            float bv = bias[col];
#pragma unroll
            for (int r = 0; r < 8; ++r) {
                int row = m0 + r + 8 * half;
                float val = acc[t][r] + bv;
                if constexpr (MODE == 1) val = fmaxf(val, 0.f);
                if constexpr (MODE == 2) {
                    val += posenc[(size_t)(row & 127) * 128 + col];
                    outF[(size_t)row * ldF + col] = val;
                }
                if constexpr (MODE == 4) {
                    if (col < outCols) outF[(size_t)row * outCols + col] = val;
                } else {
                    outB[(size_t)row * ldB16 + col] = __float2bfloat16(val);
                }
            }
        }
    }
}

// ---------------------------------------------------------------------------
// fused temporal attention: one block per (b, head). K=128, HDT=32.
// scores: 8 WMMAs/wave; register softmax; probs via LDS (C->A layout);
// V transposed into LDS; attn@V: 8 WMMAs/wave.
// ---------------------------------------------------------------------------
__global__ __launch_bounds__(256) void attn_kernel(
    const bf16* __restrict__ qkv,   // [B*K, 384] bf16
    bf16* __restrict__ outO) {      // [B*K, 128] bf16
    __shared__ bf16 vT[32 * 128];         // V^T  [d][t]     8 KB
    __shared__ bf16 probs[8 * 16 * 128];  // per-wave probs  32 KB
    const int bh = blockIdx.x;
    const int b  = bh >> 2;
    const int h  = bh & 3;
    const bf16* base = qkv + (size_t)b * 128 * 384;

    for (int idx = threadIdx.x; idx < 128 * 32; idx += 256) {
        int t = idx >> 5, d = idx & 31;
        vT[d * 128 + t] = base[(size_t)t * 384 + 256 + h * 32 + d];
    }
    __syncthreads();

    const int lane = threadIdx.x & 31;
    const int wave = threadIdx.x >> 5;
    const int l15  = lane & 15;
    const int half = lane >> 4;
    const int q0   = wave * 16;

    // scores = q @ k^T  (HDT=32 == one WMMA K)
    v8f acc[8];
#pragma unroll
    for (int t = 0; t < 8; ++t) acc[t] = zero8();
    Frag qa;
    const bf16* qrow = base + (size_t)(q0 + l15) * 384 + h * 32;
    qa.u4[0] = *(const v4u*)(qrow + 16 * half);
    qa.u4[1] = *(const v4u*)(qrow + 8 + 16 * half);
#pragma unroll
    for (int ct = 0; ct < 8; ++ct) {
        Frag kb;
        const bf16* krow = base + (size_t)(ct * 16 + l15) * 384 + 128 + h * 32 + 16 * half;
        kb.u4[0] = *(const v4u*)(krow);
        kb.u4[1] = *(const v4u*)(krow + 8);
        acc[ct] = wmma_bf16(qa, kb, acc[ct]);
    }

    // softmax over 128 keys (scale 1/sqrt(32))
    const float sc = 0.17677669529663687f;
#pragma unroll
    for (int ct = 0; ct < 8; ++ct)
#pragma unroll
        for (int r = 0; r < 8; ++r) acc[ct][r] *= sc;

    float inv[8];
#pragma unroll
    for (int r = 0; r < 8; ++r) {
        float m = -1e30f;
#pragma unroll
        for (int ct = 0; ct < 8; ++ct) m = fmaxf(m, acc[ct][r]);
#pragma unroll
        for (int mm = 1; mm < 16; mm <<= 1) m = fmaxf(m, __shfl_xor(m, mm, 32));
        float s = 0.f;
#pragma unroll
        for (int ct = 0; ct < 8; ++ct) {
            float e = __expf(acc[ct][r] - m);
            acc[ct][r] = e;
            s += e;
        }
#pragma unroll
        for (int mm = 1; mm < 16; mm <<= 1) s += __shfl_xor(s, mm, 32);
        inv[r] = 1.f / s;
    }

    // C layout -> LDS (each wave its own 16x128 region)
    bf16* myp = probs + wave * 16 * 128;
#pragma unroll
    for (int ct = 0; ct < 8; ++ct)
#pragma unroll
        for (int r = 0; r < 8; ++r)
            myp[(r + 8 * half) * 128 + ct * 16 + l15] =
                __float2bfloat16(acc[ct][r] * inv[r]);

    // out = probs @ V   (keys chunked by 32)
    v8f oacc[2];
    oacc[0] = zero8(); oacc[1] = zero8();
#pragma unroll
    for (int kc = 0; kc < 4; ++kc) {
        Frag pa;
        const bf16* prow = myp + l15 * 128 + kc * 32;
        pa.u4[0] = *(const v4u*)(prow + 16 * half);
        pa.u4[1] = *(const v4u*)(prow + 8 + 16 * half);
#pragma unroll
        for (int nt = 0; nt < 2; ++nt) {
            Frag vb;
            const bf16* vrow = vT + (nt * 16 + l15) * 128 + kc * 32 + 16 * half;
            vb.u4[0] = *(const v4u*)(vrow);
            vb.u4[1] = *(const v4u*)(vrow + 8);
            oacc[nt] = wmma_bf16(pa, vb, oacc[nt]);
        }
    }
#pragma unroll
    for (int nt = 0; nt < 2; ++nt)
#pragma unroll
        for (int r = 0; r < 8; ++r) {
            int rowT = q0 + r + 8 * half;
            int col  = h * 32 + nt * 16 + l15;
            outO[(size_t)(b * 128 + rowT) * 128 + col] = __float2bfloat16(oacc[nt][r]);
        }
}

// ---------------------------------------------------------------------------
// head LayerNorm on last timestep -> bf16
// ---------------------------------------------------------------------------
__global__ __launch_bounds__(128) void hln_kernel(
    const float* __restrict__ xF, const float* __restrict__ g,
    const float* __restrict__ bb, bf16* __restrict__ zB) {
    int b = blockIdx.x, t = threadIdx.x;
    float v = xF[((size_t)b * 128 + 127) * 128 + t];
    __shared__ float sh1[4], sh2[4];
    float sA = v;
#pragma unroll
    for (int m = 1; m < 32; m <<= 1) sA += __shfl_xor(sA, m, 32);
    if ((t & 31) == 0) sh1[t >> 5] = sA;
    __syncthreads();
    float mu = (sh1[0] + sh1[1] + sh1[2] + sh1[3]) * (1.f / 128.f);
    float d = v - mu;
    float sQ = d * d;
#pragma unroll
    for (int m = 1; m < 32; m <<= 1) sQ += __shfl_xor(sQ, m, 32);
    if ((t & 31) == 0) sh2[t >> 5] = sQ;
    __syncthreads();
    float var = (sh2[0] + sh2[1] + sh2[2] + sh2[3]) * (1.f / 128.f);
    zB[(size_t)b * 128 + t] = __float2bfloat16(d * rsqrtf(var + 1e-5f) * g[t] + bb[t]);
}

// ---------------------------------------------------------------------------
// host-side orchestration
// ---------------------------------------------------------------------------
extern "C" void kernel_launch(void* const* d_in, const int* in_sizes, int n_in,
                              void* d_out, int out_size, void* d_ws, size_t ws_size,
                              hipStream_t stream) {
    const float* prices     = (const float*)d_in[0];
    const float* gdelt      = (const float*)d_in[1];
    const float* a_static   = (const float*)d_in[2];
    const float* wg         = (const float*)d_in[3];
    const float* bg         = (const float*)d_in[4];
    const float* we         = (const float*)d_in[5];
    const float* be         = (const float*)d_in[6];
    const float* gat_W      = (const float*)d_in[7];
    const float* gat_a      = (const float*)d_in[8];
    const float* gat_out_W  = (const float*)d_in[9];
    const float* gat_out_b  = (const float*)d_in[10];
    const float* seq_proj_W = (const float*)d_in[11];
    const float* seq_proj_b = (const float*)d_in[12];
    const float* pos_enc    = (const float*)d_in[13];
    const float* qkv_w      = (const float*)d_in[14];
    const float* qkv_b      = (const float*)d_in[15];
    const float* attn_ow    = (const float*)d_in[16];
    const float* attn_ob    = (const float*)d_in[17];
    const float* ln1_g      = (const float*)d_in[18];
    const float* ln1_b      = (const float*)d_in[19];
    const float* ffn_w1     = (const float*)d_in[20];
    const float* ffn_b1     = (const float*)d_in[21];
    const float* ffn_w2     = (const float*)d_in[22];
    const float* ffn_b2     = (const float*)d_in[23];
    const float* ln2_g      = (const float*)d_in[24];
    const float* ln2_b      = (const float*)d_in[25];
    const float* hln_g      = (const float*)d_in[26];
    const float* hln_b      = (const float*)d_in[27];
    const float* head_w1    = (const float*)d_in[28];
    const float* head_b1    = (const float*)d_in[29];
    const float* head_w2    = (const float*)d_in[30];
    const float* head_b2    = (const float*)d_in[31];
    float* outP = (float*)d_out;

    // ---- workspace carving (deterministic offsets) ----
    char* ws = (char*)d_ws;
    size_t off = 0;
    auto take = [&](size_t bytes) -> char* {
        char* p = ws + off;
        off += (bytes + 255) & ~(size_t)255;
        return p;
    };
    float* wsPrep = (float*)take(1040 * sizeof(float));
    float* adyn   = (float*)take((size_t)Bn * 25 * sizeof(float));
    bf16* seqT = (bf16*)take((size_t)128 * 640 * 2);
    bf16* qkvT = (bf16*)take((size_t)2 * 384 * 128 * 2);
    bf16* owT  = (bf16*)take((size_t)2 * 128 * 128 * 2);
    bf16* w1T  = (bf16*)take((size_t)2 * 512 * 128 * 2);
    bf16* w2T  = (bf16*)take((size_t)2 * 128 * 512 * 2);
    bf16* h1T  = (bf16*)take((size_t)256 * 128 * 2);
    bf16* h2T  = (bf16*)take((size_t)16 * 256 * 2);
    bf16*  Xg   = (bf16*)take((size_t)Mrows * NDflat * 2);   // reused as FFN mid
    float* xF   = (float*)take((size_t)Mrows * 128 * 4);
    bf16*  xB   = (bf16*)take((size_t)Mrows * 128 * 2);
    bf16*  qkvB = (bf16*)take((size_t)Mrows * 384 * 2);
    bf16*  oB   = (bf16*)take((size_t)Mrows * 128 * 2);
    bf16*  zB   = (bf16*)take((size_t)Bn * 128 * 2);
    bf16*  h1B  = (bf16*)take((size_t)Bn * 256 * 2);
    bf16*  fB   = Xg;  // alias: Xg dead after seq_proj GEMM

    auto wconv = [&](const float* src, bf16* dst, int R, int C, int Cpad) {
        int total = Cpad * R;
        wconv_kernel<<<(total + 255) / 256, 256, 0, stream>>>(src, dst, R, C, Cpad);
    };

    // weights -> bf16 transposed [Nc, Kc]
    wconv(seq_proj_W, seqT, 640, 128, 128);
    for (int l = 0; l < 2; ++l) {
        wconv(qkv_w  + (size_t)l * 128 * 384, qkvT + (size_t)l * 384 * 128, 128, 384, 384);
        wconv(attn_ow + (size_t)l * 128 * 128, owT + (size_t)l * 128 * 128, 128, 128, 128);
        wconv(ffn_w1 + (size_t)l * 128 * 512, w1T + (size_t)l * 512 * 128, 128, 512, 512);
        wconv(ffn_w2 + (size_t)l * 512 * 128, w2T + (size_t)l * 128 * 512, 512, 128, 128);
    }
    wconv(head_w1, h1T, 128, 256, 256);
    wconv(head_w2, h2T, 256, 5, 16);   // pad N to 16

    prep_kernel<<<1, 128, 0, stream>>>(we, be, gat_W, gat_a, gat_out_W, wsPrep);
    adyn_kernel<<<(Bn * 25 + 255) / 256, 256, 0, stream>>>(gdelt, a_static, wg, bg,
                                                           adyn, Bn * 25);
    gat_kernel<<<Mrows, 128, 0, stream>>>(prices, adyn, wsPrep, gat_out_b, Xg);

    // seq_proj + bias + pos_enc -> x (f32 + bf16)
    gemm_kernel<8, 2><<<dim3(Mrows / 128, 1), 256, 0, stream>>>(
        Xg, 640, 640, seqT, seq_proj_b, nullptr, 0, nullptr, nullptr,
        pos_enc, xF, 128, xB, 128, 0);

    for (int l = 0; l < 2; ++l) {
        // qkv
        gemm_kernel<8, 0><<<dim3(Mrows / 128, 3), 256, 0, stream>>>(
            xB, 128, 128, qkvT + (size_t)l * 384 * 128, qkv_b + l * 384,
            nullptr, 0, nullptr, nullptr, nullptr, nullptr, 0, qkvB, 384, 0);
        // temporal attention
        attn_kernel<<<Bn * 4, 256, 0, stream>>>(qkvB, oB);
        // attn out proj + residual + LN1
        gemm_kernel<8, 3><<<dim3(Mrows / 128, 1), 256, 0, stream>>>(
            oB, 128, 128, owT + (size_t)l * 128 * 128, attn_ob + l * 128,
            xF, 128, ln1_g + l * 128, ln1_b + l * 128, nullptr,
            xF, 128, xB, 128, 0);
        // FFN1 (relu)
        gemm_kernel<8, 1><<<dim3(Mrows / 128, 4), 256, 0, stream>>>(
            xB, 128, 128, w1T + (size_t)l * 512 * 128, ffn_b1 + l * 512,
            nullptr, 0, nullptr, nullptr, nullptr, nullptr, 0, fB, 512, 0);
        // FFN2 + residual + LN2
        gemm_kernel<8, 3><<<dim3(Mrows / 128, 1), 256, 0, stream>>>(
            fB, 512, 512, w2T + (size_t)l * 128 * 512, ffn_b2 + l * 128,
            xF, 128, ln2_g + l * 128, ln2_b + l * 128, nullptr,
            xF, 128, xB, 128, 0);
    }

    // head
    hln_kernel<<<Bn, 128, 0, stream>>>(xF, hln_g, hln_b, zB);
    gemm_kernel<8, 1><<<dim3(Bn / 128, 2), 256, 0, stream>>>(
        zB, 128, 128, h1T, head_b1, nullptr, 0, nullptr, nullptr, nullptr,
        nullptr, 0, h1B, 256, 0);
    gemm_kernel<1, 4><<<dim3(Bn / 128, 1), 256, 0, stream>>>(
        h1B, 256, 256, h2T, head_b2, nullptr, 0, nullptr, nullptr, nullptr,
        outP, 0, nullptr, 0, 5);
}
